// SlotAttention_76278619177346
// MI455X (gfx1250) — matrix operand order repaired and checked by reading, not compiled, not run on hardware
//
#include <hip/hip_runtime.h>
#include <hip/hip_bf16.h>
#include <math.h>

#define NH  16
#define HD  128
#define BSZ 8
#define SSZ 8192
#define HSZ 2048
#define SL  16

typedef __attribute__((ext_vector_type(16))) __bf16 v16bf;
typedef __attribute__((ext_vector_type(8)))  float  v8f;
typedef __attribute__((ext_vector_type(4)))  unsigned int v4u;
typedef __attribute__((ext_vector_type(8)))  int v8i;
typedef __attribute__((ext_vector_type(4)))  int v4i;

struct B32x8 { uint4 lo, hi; };

static __device__ __forceinline__ v16bf as_v16bf(uint4 lo, uint4 hi) {
  B32x8 t{lo, hi};
  return __builtin_bit_cast(v16bf, t);
}

// f32 -> bf16 (round-to-nearest-even)
static __device__ __forceinline__ unsigned f2bf1(float f) {
  unsigned u = __builtin_bit_cast(unsigned, f);
  return (u + 0x7FFFu + ((u >> 16) & 1u)) >> 16;
}
static __device__ __forceinline__ unsigned packbf(float lo, float hi) {
  return f2bf1(lo) | (f2bf1(hi) << 16);
}
static __device__ __forceinline__ uint4 pack8(float4 a, float4 b) {
  return make_uint4(packbf(a.x, a.y), packbf(a.z, a.w),
                    packbf(b.x, b.y), packbf(b.z, b.w));
}

static __device__ __forceinline__ v8f wmma_bf16(v16bf a, v16bf b, v8f c) {
  return __builtin_amdgcn_wmma_f32_16x16x32_bf16(false, a, false, b, (short)0, c,
                                                 false, false);
}

// LDS byte offset of a __shared__ object (flat LDS addresses carry the
// workgroup-relative offset in addr[31:0] per CDNA5 aperture rules).
static __device__ __forceinline__ unsigned lds_offset(const void* p) {
  return (unsigned)(unsigned long long)(uintptr_t)p;
}

// ---------------------------------------------------------------------------
// Tensor Data Mover: async 2D f32 tile load (tile_d0 x tile_d1 elements,
// row stride stride0 elements) from global to LDS. Descriptor per
// CDNA5 ISA. One DMA per call (issue per-wave),
// completion tracked with TENSORcnt.
// ---------------------------------------------------------------------------
static __device__ __forceinline__ void tdm_load_2d_f32(
    unsigned lds_off, const void* gptr,
    unsigned tensor_d0, unsigned tensor_d1,
    unsigned tile_d0, unsigned tile_d1, unsigned stride0) {
  unsigned long long ga = (unsigned long long)(uintptr_t)gptr;
  v4u g0;
  g0[0] = 1u;                                        // count=1 (valid), user mode
  g0[1] = lds_off;                                   // lds_addr
  g0[2] = (unsigned)(ga & 0xFFFFFFFFu);              // global_addr[31:0]
  g0[3] = (unsigned)((ga >> 32) & 0x01FFFFFFu)       // global_addr[56:32]
          | (2u << 30);                              // type=2 ("image")
  v8i g1;
  g1[0] = (int)(2u << 16);                           // data_size=4B, no flags
  g1[1] = (int)((tensor_d0 & 0xFFFFu) << 16);        // tensor_dim0[15:0]
  g1[2] = (int)(((tensor_d0 >> 16) & 0xFFFFu) |
                ((tensor_d1 & 0xFFFFu) << 16));      // dim0 hi | dim1 lo
  g1[3] = (int)(((tensor_d1 >> 16) & 0xFFFFu) |
                (tile_d0 << 16));                    // dim1 hi | tile_dim0
  g1[4] = (int)tile_d1;                              // tile_dim1 (tile_dim2=0)
  g1[5] = (int)stride0;                              // tensor_dim0_stride[31:0]
  g1[6] = 0;                                         // stride0 hi | stride1 lo
  g1[7] = 0;
  v4i z4 = {0, 0, 0, 0};
#if __clang_major__ >= 23
  v8i z8 = {0, 0, 0, 0, 0, 0, 0, 0};
  __builtin_amdgcn_tensor_load_to_lds(g0, g1, z4, z4, z8, 0);
#else
  __builtin_amdgcn_tensor_load_to_lds(g0, g1, z4, z4, 0);
#endif
}

// ---------------------------------------------------------------------------
// Generic small GEMM: C[M,N] = A[M,K] (f32) @ W[N,K]^T (f32), bf16 WMMA.
// grid = (N/16, M/16), block = 32 (one wave).
// ---------------------------------------------------------------------------
__global__ __launch_bounds__(32)
void proj_gemm_kernel(const float* __restrict__ A, const float* __restrict__ W,
                      float* __restrict__ C, int K, int N) {
  const int lane = threadIdx.x & 31;
  const int n0 = blockIdx.x * 16;
  const int m0 = blockIdx.y * 16;
  const float* arow = A + (size_t)(m0 + (lane & 15)) * K + (lane >> 4) * 16;
  const float* brow = W + (size_t)(n0 + (lane & 15)) * K + (lane >> 4) * 16;
  v8f acc = {};
  for (int k = 0; k < K; k += 32) {
    if (k + 32 < K) { __builtin_prefetch(arow + k + 32, 0, 1);
                      __builtin_prefetch(brow + k + 32, 0, 1); }
    float4 a0 = *(const float4*)(arow + k);
    float4 a1 = *(const float4*)(arow + k + 4);
    float4 a2 = *(const float4*)(arow + k + 8);
    float4 a3 = *(const float4*)(arow + k + 12);
    float4 b0 = *(const float4*)(brow + k);
    float4 b1 = *(const float4*)(brow + k + 4);
    float4 b2 = *(const float4*)(brow + k + 8);
    float4 b3 = *(const float4*)(brow + k + 12);
    acc = wmma_bf16(as_v16bf(pack8(a0, a1), pack8(a2, a3)),
                    as_v16bf(pack8(b0, b1), pack8(b2, b3)), acc);
  }
  for (int r = 0; r < 8; ++r)
    C[(size_t)(m0 + (lane >> 4) * 8 + r) * N + n0 + (lane & 15)] = acc[r];
}

// ---------------------------------------------------------------------------
// Fused K/V projection + RoPE(K) + bf16 spill. Tiles move via the Tensor
// Data Mover (async DMA, TENSORcnt) instead of per-thread loads+ds_stores.
//   N dimension: [0,2048)=K-features, [2048,4096)=V-features.
//   Khat: bf16 [b,h,s,d]   Vt: bf16 [b,h,d,s]  (transposed through LDS)
// grid = (4096/64, 65536/64), block = 256 (8 waves, 4(M)x2(N) layout).
// ---------------------------------------------------------------------------
__global__ __launch_bounds__(256)
void kv_proj_rope_kernel(const float* __restrict__ tokens,
                         const float* __restrict__ Wk,
                         const float* __restrict__ Wv,
                         unsigned short* __restrict__ khat,
                         unsigned short* __restrict__ vt) {
  __shared__ float lAf[64 * 32];   // tokens tile  (rows x 32 K), f32 via TDM
  __shared__ float lBf[64 * 32];   // weights tile (feat x 32 K), f32 via TDM
  __shared__ float lT[8][2][16 * 16];

  const int tid  = threadIdx.x;
  const int lane = tid & 31;
  const int w    = tid >> 5;
  const int wm   = (w & 3) * 16;
  const int wn   = (w >> 2) * 32;
  const int n0   = blockIdx.x * 64;
  const int m0   = blockIdx.y * 64;
  const bool isV = (n0 >= HSZ);

  const float* abase = tokens + (size_t)m0 * HSZ;
  const float* wbase = isV ? (Wv + (size_t)(n0 - HSZ) * HSZ)
                           : (Wk + (size_t)n0 * HSZ);
  const unsigned ldsA = lds_offset(lAf);
  const unsigned ldsB = lds_offset(lBf);

  v8f acc0 = {}, acc1 = {};

  for (int k = 0; k < HSZ; k += 32) {
    __syncthreads();                       // previous iteration's reads done
    if (w == 0) {                          // one wave drives both tile DMAs
      tdm_load_2d_f32(ldsA, abase + k, HSZ, BSZ * SSZ, 32, 64, HSZ);
      tdm_load_2d_f32(ldsB, wbase + k, HSZ, HSZ,       32, 64, HSZ);
      __builtin_amdgcn_s_wait_tensorcnt(0);
    }
    __syncthreads();                       // tiles visible to all waves

    const float* pa = lAf + (wm + (lane & 15)) * 32 + (lane >> 4) * 16;
    float4 a0 = *(const float4*)(pa);
    float4 a1 = *(const float4*)(pa + 4);
    float4 a2 = *(const float4*)(pa + 8);
    float4 a3 = *(const float4*)(pa + 12);
    v16bf afrag = as_v16bf(pack8(a0, a1), pack8(a2, a3));

    const float* pb0 = lBf + (wn + (lane & 15)) * 32 + (lane >> 4) * 16;
    float4 c0 = *(const float4*)(pb0);
    float4 c1 = *(const float4*)(pb0 + 4);
    float4 c2 = *(const float4*)(pb0 + 8);
    float4 c3 = *(const float4*)(pb0 + 12);
    const float* pb1 = pb0 + 16 * 32;
    float4 d0 = *(const float4*)(pb1);
    float4 d1 = *(const float4*)(pb1 + 4);
    float4 d2 = *(const float4*)(pb1 + 8);
    float4 d3 = *(const float4*)(pb1 + 12);
    acc0 = wmma_bf16(afrag, as_v16bf(pack8(c0, c1), pack8(c2, c3)), acc0);
    acc1 = wmma_bf16(afrag, as_v16bf(pack8(d0, d1), pack8(d2, d3)), acc1);
  }

  const int mrow = (lane >> 4) * 8;
  if (!isV) {
    // RoPE + store Khat[b,h,s,d]. rotate_half pairs (2i,2i+1) = lane parity.
    for (int t = 0; t < 2; ++t) {
      v8f& acc = t ? acc1 : acc0;
      int gn = n0 + wn + t * 16 + (lane & 15);
      int head = gn >> 7;
      int d = gn & 127;
      float invf = __expf(-(float)(d & 63) * (9.210340371976184f / 64.0f));
      for (int r = 0; r < 8; ++r) {
        int grow = m0 + wm + mrow + r;
        int b = grow >> 13;
        int s = grow & (SSZ - 1);
        float sn, cs;
        __sincosf((float)s * invf, &sn, &cs);
        float x = acc[r];
        float part = __shfl_xor(x, 1);
        float rot = (lane & 1) ? part : -part;
        float kh = x * cs + rot * sn;
        khat[((size_t)(b * NH + head) * SSZ + s) * HD + d] = (unsigned short)f2bf1(kh);
      }
    }
  } else {
    // transpose V tiles through LDS -> packed 32B stores to Vt[b,h,d,s]
    for (int t = 0; t < 2; ++t) {
      v8f& acc = t ? acc1 : acc0;
      for (int r = 0; r < 8; ++r)
        lT[w][t][(mrow + r) * 16 + (lane & 15)] = acc[r];
    }
    __syncthreads();
    int t   = lane >> 4;
    int col = lane & 15;
    int gn  = n0 - HSZ + wn + t * 16 + col;
    int head = gn >> 7;
    int d    = gn & 127;
    int grow0 = m0 + wm;
    int b  = grow0 >> 13;
    int s0 = grow0 & (SSZ - 1);
    unsigned pk[8];
    for (int i = 0; i < 8; ++i)
      pk[i] = packbf(lT[w][t][(2 * i) * 16 + col], lT[w][t][(2 * i + 1) * 16 + col]);
    uint4* dst = (uint4*)(vt + ((size_t)(b * NH + head) * HD + d) * SSZ + s0);
    dst[0] = make_uint4(pk[0], pk[1], pk[2], pk[3]);
    dst[1] = make_uint4(pk[4], pk[5], pk[6], pk[7]);
  }
}

// ---------------------------------------------------------------------------
// Streaming slot attention per (b,h): flash-style online softmax.
// block = 128 (4 waves), each wave owns 2048 keys; combine via LDS.
// ctxws: f32 [b*16+slot][2048] with head slice at h*128.
// ---------------------------------------------------------------------------
__global__ __launch_bounds__(128)
void slot_attention_kernel(const unsigned short* __restrict__ khat,
                           const unsigned short* __restrict__ vt,
                           const float* __restrict__ qws,
                           const int* __restrict__ mask,
                           float* __restrict__ ctxws) {
  __shared__ unsigned short lP[4][16][32];
  __shared__ float cm[4][16], cl[4][16];
  __shared__ float cctx[4][SL][HD];

  const int h = blockIdx.x, b = blockIdx.y;
  const int tid = threadIdx.x, lane = tid & 31, w = tid >> 5;
  const unsigned short* kh = khat + (size_t)(b * NH + h) * SSZ * HD;
  const unsigned short* vh = vt   + (size_t)(b * NH + h) * HD * SSZ;
  const int* mk = mask + b * SSZ;

  // q fragments (A-matrix: lane&15 = slot, K = head-dim)
  v16bf qf[4];
  for (int c = 0; c < 4; ++c) {
    const float* qp = qws + (size_t)(lane & 15) * HSZ + h * HD + c * 32 + (lane >> 4) * 16;
    float4 q0 = *(const float4*)qp;
    float4 q1 = *(const float4*)(qp + 4);
    float4 q2 = *(const float4*)(qp + 8);
    float4 q3 = *(const float4*)(qp + 12);
    qf[c] = as_v16bf(pack8(q0, q1), pack8(q2, q3));
  }

  v8f zero = {};
  v8f ctx[8];
  for (int n = 0; n < 8; ++n) ctx[n] = zero;
  float m8[8], l8[8];
  for (int i = 0; i < 8; ++i) { m8[i] = -__builtin_huge_valf(); l8[i] = 0.f; }
  const float scale = 0.08838834764831845f;  // 1/sqrt(128)

  for (int ch = 0; ch < 64; ++ch) {
    const int kb = w * 2048 + ch * 32;
    if (ch + 1 < 64) {
      __builtin_prefetch(kh + (size_t)(kb + 32 + (lane & 15)) * HD, 0, 1);
      __builtin_prefetch(vh + (size_t)(lane & 15) * SSZ + kb + 32, 0, 1);
    }
    v8f sc0 = zero, sc1 = zero;
    for (int c = 0; c < 4; ++c) {
      const uint4* p0 = (const uint4*)(kh + (size_t)(kb + (lane & 15)) * HD + c * 32 + (lane >> 4) * 16);
      const uint4* p1 = (const uint4*)(kh + (size_t)(kb + 16 + (lane & 15)) * HD + c * 32 + (lane >> 4) * 16);
      sc0 = wmma_bf16(qf[c], as_v16bf(p0[0], p0[1]), sc0);
      sc1 = wmma_bf16(qf[c], as_v16bf(p1[0], p1[1]), sc1);
    }
    const int mv0 = mk[kb + (lane & 15)];
    const int mv1 = mk[kb + 16 + (lane & 15)];
    float p0[8], p1[8], alpha[8];
    for (int i = 0; i < 8; ++i) {
      float s0 = mv0 ? sc0[i] * scale : -3.0e38f;
      float s1 = mv1 ? sc1[i] * scale : -3.0e38f;
      float mx = fmaxf(s0, s1);
      for (int x = 1; x < 16; x <<= 1) mx = fmaxf(mx, __shfl_xor(mx, x));
      float mnew = fmaxf(m8[i], mx);
      alpha[i] = __expf(m8[i] - mnew);
      m8[i] = mnew;
      p0[i] = __expf(s0 - mnew);
      p1[i] = __expf(s1 - mnew);
      float ls = p0[i] + p1[i];
      for (int x = 1; x < 16; x <<= 1) ls += __shfl_xor(ls, x);
      l8[i] = l8[i] * alpha[i] + ls;
    }
    for (int n = 0; n < 8; ++n)
      for (int i = 0; i < 8; ++i) ctx[n][i] *= alpha[i];
    // P -> per-wave LDS region in A-fragment layout
    for (int i = 0; i < 8; ++i) {
      int slot = (lane >> 4) * 8 + i;
      lP[w][slot][lane & 15]        = (unsigned short)f2bf1(p0[i]);
      lP[w][slot][16 + (lane & 15)] = (unsigned short)f2bf1(p1[i]);
    }
    const uint4* pp = (const uint4*)(&lP[w][lane & 15][(lane >> 4) * 16]);
    v16bf pfrag = as_v16bf(pp[0], pp[1]);
    for (int n = 0; n < 8; ++n) {
      const uint4* pv = (const uint4*)(vh + (size_t)(n * 16 + (lane & 15)) * SSZ + kb + (lane >> 4) * 16);
      ctx[n] = wmma_bf16(pfrag, as_v16bf(pv[0], pv[1]), ctx[n]);
    }
  }

  // combine 4 waves
  if ((lane & 15) == 0)
    for (int i = 0; i < 8; ++i) {
      int slot = (lane >> 4) * 8 + i;
      cm[w][slot] = m8[i];
      cl[w][slot] = l8[i];
    }
  for (int n = 0; n < 8; ++n)
    for (int i = 0; i < 8; ++i)
      cctx[w][(lane >> 4) * 8 + i][n * 16 + (lane & 15)] = ctx[n][i];
  __syncthreads();

  {
    const int slot  = tid >> 3;          // 16 elems per thread, same slot
    const int dbase = (tid & 7) * 16;
    float M = cm[0][slot];
    for (int ww = 1; ww < 4; ++ww) M = fmaxf(M, cm[ww][slot]);
    float sc[4], L = 0.f;
    for (int ww = 0; ww < 4; ++ww) { sc[ww] = __expf(cm[ww][slot] - M); L += cl[ww][slot] * sc[ww]; }
    float invL = 1.0f / L;
    float* out = ctxws + (size_t)(b * SL + slot) * HSZ + h * HD;
    for (int j = 0; j < 16; ++j) {
      int d = dbase + j;
      float v = 0.f;
      for (int ww = 0; ww < 4; ++ww) v += cctx[ww][slot][d] * sc[ww];
      out[d] = v * invL;
    }
  }
}

// ---------------------------------------------------------------------------
extern "C" void kernel_launch(void* const* d_in, const int* in_sizes, int n_in,
                              void* d_out, int out_size, void* d_ws, size_t ws_size,
                              hipStream_t stream) {
  (void)in_sizes; (void)n_in; (void)out_size; (void)ws_size;
  const float* tokens = (const float*)d_in[0];
  const int*   mask   = (const int*)d_in[1];
  const float* slotq  = (const float*)d_in[2];
  const float* Wq     = (const float*)d_in[3];
  const float* Wk     = (const float*)d_in[4];
  const float* Wv     = (const float*)d_in[5];
  const float* Wo     = (const float*)d_in[6];

  char* ws = (char*)d_ws;
  float* q_ws   = (float*)(ws);                                  // 16*2048*4
  float* ctx_ws = (float*)(ws + 131072);                         // 128*2048*4
  unsigned short* khat = (unsigned short*)(ws + (size_t)(2u << 20));
  unsigned short* vtp  = (unsigned short*)(ws + (size_t)(2u << 20) +
                                           (size_t)BSZ * NH * SSZ * HD * 2);

  // q = slot_queries @ Wq^T
  proj_gemm_kernel<<<dim3(HSZ / 16, SL / 16), 32, 0, stream>>>(slotq, Wq, q_ws, HSZ, HSZ);
  // fused K/V projection + RoPE, bf16 spill (tokens read once, tiles via TDM)
  kv_proj_rope_kernel<<<dim3(2 * HSZ / 64, (BSZ * SSZ) / 64), 256, 0, stream>>>(
      tokens, Wk, Wv, khat, vtp);
  // streaming attention with online softmax
  slot_attention_kernel<<<dim3(NH, BSZ), 128, 0, stream>>>(khat, vtp, q_ws, mask, ctx_ws);
  // out = ctx @ Wo^T
  proj_gemm_kernel<<<dim3(HSZ / 16, (BSZ * SL) / 16), 32, 0, stream>>>(
      ctx_ws, Wo, (float*)d_out, HSZ, HSZ);
}
